// snn_2_55499567399222
// MI455X (gfx1250) — matrix operand order repaired and checked by reading, not compiled, not run on hardware
//
#include <hip/hip_runtime.h>
#include <hip/hip_bf16.h>

typedef _Float16 v8h  __attribute__((ext_vector_type(8)));
typedef _Float16 v16h __attribute__((ext_vector_type(16)));
typedef float    v8f  __attribute__((ext_vector_type(8)));

#define LDST 40   // LDS row stride in halves (80B): 16B-aligned, conflict-free frag loads

// Async global -> LDS copy of 16B per lane (CDNA5 GLOBAL_LOAD_ASYNC_TO_LDS_B128,
// tracked by ASYNCcnt). lds_off is the allocation-relative LDS byte offset.
__device__ __forceinline__ void async_copy_b128(unsigned int lds_off, const void* g) {
    asm volatile("global_load_async_to_lds_b128 %0, %1, off"
                 :: "v"(lds_off), "v"((unsigned long long)(uintptr_t)g)
                 : "memory");
}

// ---------------------------------------------------------------------------
// avg_pool3d (3,2,2)/(2,2,2) over [B=32,C=4,232,64,64] -> f16 feat [8,1792,1024]
// feat index = ((t*32+b)*56 + (c*14+tp))*1024 + hp*32+wp
// ---------------------------------------------------------------------------
__global__ __launch_bounds__(256) void pool_kernel(const float* __restrict__ x,
                                                   _Float16* __restrict__ featH) {
    const int idx = blockIdx.x * 256 + threadIdx.x;        // exactly 14,680,064 threads
    const int d   = idx & 1023;
    const int tmp = idx >> 10;
    const int r   = tmp % 56;
    const int tb  = tmp / 56;
    const int b   = tb & 31;
    const int t   = tb >> 5;
    const int c   = r / 14;
    const int tp  = r % 14;
    const int hp  = d >> 5;
    const int wp  = d & 31;
    const int dd  = t * 29 + tp * 2;
    const int hh  = hp * 2;
    const int ww  = wp * 2;
    const size_t base = ((size_t)(b * 4 + c) * 232 + dd) * 4096 + hh * 64 + ww;
    float s = 0.f;
#pragma unroll
    for (int kd = 0; kd < 3; ++kd) {
        const float* p = x + base + (size_t)kd * 4096;
        s += p[0] + p[1] + p[64] + p[65];
    }
    featH[idx] = (_Float16)(s * (1.0f / 12.0f));
}

// ---------------------------------------------------------------------------
// W [K,N] fp32 -> Wt [N,K] f16
// ---------------------------------------------------------------------------
__global__ __launch_bounds__(256) void w_transpose_kernel(const float* __restrict__ W,
                                                          _Float16* __restrict__ Wt,
                                                          int K, int N) {
    const int idx = blockIdx.x * 256 + threadIdx.x;
    if (idx >= K * N) return;
    const int k = idx / N;
    const int n = idx - k * N;
    Wt[(size_t)n * K + k] = (_Float16)W[idx];
}

__global__ void zero_kernel(float* __restrict__ p, int n) {
    for (int i = blockIdx.x * blockDim.x + threadIdx.x; i < n; i += gridDim.x * blockDim.x)
        p[i] = 0.f;
}

// ---------------------------------------------------------------------------
// Fused GEMM + LIF:  cur = A(f16,MxK) @ Bt^T(f16,NxK) + bias
//   reset   = (mem_old > thr) * thr
//   mem_new = clip(beta,0,1)*mem_old + cur - reset
//   spk     = (mem_new - thr) > 0  -> f16 0/1
// Block tile 128x128, 8 waves, wave tile 32x64 (2x4 wmma), K step 32.
// Double-buffered LDS, async global->LDS staging one K-tile ahead (ASYNCcnt).
// ---------------------------------------------------------------------------
__global__ __launch_bounds__(256)
void gemm_lif_kernel(const _Float16* __restrict__ A, const _Float16* __restrict__ Bt,
                     const float* __restrict__ bias, float* __restrict__ mem,
                     _Float16* __restrict__ spk,
                     const float* __restrict__ betas, int bidx,
                     const float* __restrict__ thrs, int tidx,
                     int M, int N, int K) {
    __shared__ _Float16 ldsA[2][128 * LDST];
    __shared__ _Float16 ldsB[2][128 * LDST];

    const int m0   = blockIdx.x * 128;
    const int n0   = blockIdx.y * 128;
    const int tid  = threadIdx.x;
    const int lane = tid & 31;
    const int wave = tid >> 5;
    const int wm   = wave & 3;       // M sub-tile: rows wm*32 .. +31
    const int wn   = wave >> 2;      // N sub-tile: cols wn*64 .. +63
    const int lr   = lane & 15;
    const int hi   = lane >> 4;      // 0: lanes 0-15, 1: lanes 16-31

    float beta = betas[bidx];
    beta = fminf(fmaxf(beta, 0.f), 1.f);
    const float thr = thrs[tidx];

    const int ldrow  = tid >> 1;           // 0..127
    const int ldhalf = (tid & 1) * 16;     // 0 or 16 halves

    const _Float16* gA = A  + (size_t)(m0 + ldrow) * K + ldhalf;
    const _Float16* gB = Bt + (size_t)(n0 + ldrow) * K + ldhalf;

    // allocation-relative LDS byte offsets for this thread's staging slots
    unsigned int sAoff[2], sBoff[2];
#pragma unroll
    for (int bfi = 0; bfi < 2; ++bfi) {
        sAoff[bfi] = (unsigned int)(uintptr_t)&ldsA[bfi][ldrow * LDST + ldhalf];
        sBoff[bfi] = (unsigned int)(uintptr_t)&ldsB[bfi][ldrow * LDST + ldhalf];
    }

    v8f acc[2][4] = {};

    // prologue: stage K-tile 0 into buffer 0 (4 async ops per wave)
    async_copy_b128(sAoff[0],      gA);
    async_copy_b128(sAoff[0] + 16, gA + 8);
    async_copy_b128(sBoff[0],      gB);
    async_copy_b128(sBoff[0] + 16, gB + 8);
    gA += 32; gB += 32;

    const int nsteps = K >> 5;
    for (int i = 0; i < nsteps; ++i) {
        const int cur = i & 1;
        if (i + 1 < nsteps) {
            const int nxt = cur ^ 1;
            async_copy_b128(sAoff[nxt],      gA);
            async_copy_b128(sAoff[nxt] + 16, gA + 8);
            async_copy_b128(sBoff[nxt],      gB);
            async_copy_b128(sBoff[nxt] + 16, gB + 8);
            gA += 32; gB += 32;
            // <=4 outstanding: the 4 just-issued; all older (current tile) complete
            asm volatile("s_wait_asynccnt 0x4" ::: "memory");
        } else {
            asm volatile("s_wait_asynccnt 0x0" ::: "memory");
        }
        __syncthreads();   // all waves' async writes for 'cur' visible

        // A fragments: lane<16 -> K {0-7,16-23}; lane>=16 -> K {8-15,24-31}; row M=lr
        v16h afrag[2];
#pragma unroll
        for (int am = 0; am < 2; ++am) {
            const _Float16* p = &ldsA[cur][(wm * 32 + am * 16 + lr) * LDST + hi * 8];
            v8h lo = *(const v8h*)p;
            v8h up = *(const v8h*)(p + 16);
#pragma unroll
            for (int q = 0; q < 8; ++q) { afrag[am][q] = lo[q]; afrag[am][q + 8] = up[q]; }
        }
        // B fragments: lane<16 -> K {0-7,16-23} at N=lr; lane>=16 -> K {8-15,24-31}
        v16h bfrag[4];
#pragma unroll
        for (int bn = 0; bn < 4; ++bn) {
            const _Float16* p = &ldsB[cur][(wn * 64 + bn * 16 + lr) * LDST + hi * 8];
            v8h lo = *(const v8h*)p;
            v8h up = *(const v8h*)(p + 16);
#pragma unroll
            for (int q = 0; q < 8; ++q) { bfrag[bn][q] = lo[q]; bfrag[bn][q + 8] = up[q]; }
        }

#pragma unroll
        for (int am = 0; am < 2; ++am)
#pragma unroll
            for (int bn = 0; bn < 4; ++bn)
                acc[am][bn] = __builtin_amdgcn_wmma_f32_16x16x32_f16(
                    false, afrag[am], false, bfrag[bn],
                    (short)0, acc[am][bn], false, false);

        __syncthreads();   // all waves done reading 'cur' before it is re-staged
    }

    // LIF epilogue. D layout: lane<16 -> row v, lane>=16 -> row 8+v; col = lr.
#pragma unroll
    for (int am = 0; am < 2; ++am) {
#pragma unroll
        for (int bn = 0; bn < 4; ++bn) {
            const int col = n0 + wn * 64 + bn * 16 + lr;
            const float bval = bias[col];
#pragma unroll
            for (int v = 0; v < 8; ++v) {
                const int row = m0 + wm * 32 + am * 16 + hi * 8 + v;
                const size_t idx = (size_t)row * N + col;
                const float cur  = acc[am][bn][v] + bval;
                const float mold = mem[idx];
                const float mnew = beta * mold + cur - ((mold > thr) ? thr : 0.f);
                mem[idx] = mnew;
                spk[idx] = (mnew > thr) ? (_Float16)1.0f : (_Float16)0.0f;
            }
        }
    }
}

// ---------------------------------------------------------------------------
// Output layer: [32 x 57344] spikes @ [57344 x 4] + LIF (thr = 1.0)
// one block per batch row
// ---------------------------------------------------------------------------
__global__ __launch_bounds__(256)
void out_lif_kernel(const _Float16* __restrict__ spk2, const float* __restrict__ Wout,
                    const float* __restrict__ bout, float* __restrict__ m_out,
                    const float* __restrict__ betas, float* __restrict__ out, int t) {
    __shared__ float red[256 * 4];
    const int b   = blockIdx.x;
    const int tid = threadIdx.x;
    const _Float16* s = spk2 + (size_t)b * 57344;
    float a0 = 0.f, a1 = 0.f, a2 = 0.f, a3 = 0.f;
    for (int f = tid; f < 57344; f += 256) {
        const float sv = (float)s[f];
        const float4 w = *(const float4*)(Wout + (size_t)f * 4);
        a0 += sv * w.x; a1 += sv * w.y; a2 += sv * w.z; a3 += sv * w.w;
    }
    red[tid * 4 + 0] = a0; red[tid * 4 + 1] = a1;
    red[tid * 4 + 2] = a2; red[tid * 4 + 3] = a3;
    __syncthreads();
    for (int sft = 128; sft > 0; sft >>= 1) {
        if (tid < sft) {
#pragma unroll
            for (int j = 0; j < 4; ++j)
                red[tid * 4 + j] += red[(tid + sft) * 4 + j];
        }
        __syncthreads();
    }
    if (tid == 0) {
        float beta = betas[3];
        beta = fminf(fmaxf(beta, 0.f), 1.f);
#pragma unroll
        for (int cls = 0; cls < 4; ++cls) {
            const float cur  = red[cls] + bout[cls];
            const float mold = m_out[b * 4 + cls];
            const float mnew = beta * mold + cur - ((mold > 1.0f) ? 1.0f : 0.f);
            m_out[b * 4 + cls] = mnew;
            out[t * 128 + b * 4 + cls] = (mnew > 1.0f) ? 1.0f : 0.0f;
        }
    }
}

// ---------------------------------------------------------------------------
extern "C" void kernel_launch(void* const* d_in, const int* in_sizes, int n_in,
                              void* d_out, int out_size, void* d_ws, size_t ws_size,
                              hipStream_t stream) {
    const float* x     = (const float*)d_in[0];
    const float* W_in  = (const float*)d_in[1];
    const float* b_in  = (const float*)d_in[2];
    const float* W_h1  = (const float*)d_in[3];
    const float* b_h1  = (const float*)d_in[4];
    const float* W_h2  = (const float*)d_in[5];
    const float* b_h2  = (const float*)d_in[6];
    const float* W_out = (const float*)d_in[7];
    const float* b_out = (const float*)d_in[8];
    const float* betas = (const float*)d_in[9];
    const float* thrs  = (const float*)d_in[10];
    float* out = (float*)d_out;

    const int M = 1792;                 // B(32) * R(56)
    char* ws = (char*)d_ws;
    size_t off = 0;
    auto take = [&](size_t bytes) { size_t o = off; off += (bytes + 255) & ~(size_t)255; return o; };

    _Float16* featH = (_Float16*)(ws + take((size_t)8 * M * 1024 * 2));   // 29.4 MB
    _Float16* WinT  = (_Float16*)(ws + take((size_t)2048 * 1024 * 2));    // [N=2048][K=1024]
    _Float16* Wh1T  = (_Float16*)(ws + take((size_t)2048 * 2048 * 2));    // [2048][2048]
    _Float16* Wh2T  = (_Float16*)(ws + take((size_t)1024 * 2048 * 2));    // [1024][2048]
    float*    m_in  = (float*)   (ws + take((size_t)M * 2048 * 4));
    float*    m_h1  = (float*)   (ws + take((size_t)M * 2048 * 4));
    float*    m_h2  = (float*)   (ws + take((size_t)M * 1024 * 4));
    float*    m_out = (float*)   (ws + take((size_t)128 * 4));
    _Float16* spkin = (_Float16*)(ws + take((size_t)M * 2048 * 2));
    _Float16* spk1  = (_Float16*)(ws + take((size_t)M * 2048 * 2));
    _Float16* spk2  = (_Float16*)(ws + take((size_t)M * 1024 * 2));

    // features + weight prep
    pool_kernel<<<57344, 256, 0, stream>>>(x, featH);
    w_transpose_kernel<<<(1024 * 2048 + 255) / 256, 256, 0, stream>>>(W_in, WinT, 1024, 2048);
    w_transpose_kernel<<<(2048 * 2048 + 255) / 256, 256, 0, stream>>>(W_h1, Wh1T, 2048, 2048);
    w_transpose_kernel<<<(2048 * 1024 + 255) / 256, 256, 0, stream>>>(W_h2, Wh2T, 2048, 1024);

    // zero membrane state (m_in..m_out contiguous, all sizes multiple of 256B)
    const int nzero = (M * 2048 + M * 2048 + M * 1024 + 128);
    zero_kernel<<<2048, 256, 0, stream>>>(m_in, nzero);

    for (int t = 0; t < 8; ++t) {
        gemm_lif_kernel<<<dim3(14, 16), 256, 0, stream>>>(
            featH + (size_t)t * M * 1024, WinT, b_in, m_in, spkin,
            betas, 0, thrs, 0, M, 2048, 1024);
        gemm_lif_kernel<<<dim3(14, 16), 256, 0, stream>>>(
            spkin, Wh1T, b_h1, m_h1, spk1,
            betas, 1, thrs, 1, M, 2048, 2048);
        gemm_lif_kernel<<<dim3(14, 8), 256, 0, stream>>>(
            spk1, Wh2T, b_h2, m_h2, spk2,
            betas, 2, thrs, 2, M, 1024, 2048);
        out_lif_kernel<<<32, 256, 0, stream>>>(spk2, W_out, b_out, m_out, betas, out, t);
    }
}
